// DecoderLSTM_46720654246140
// MI455X (gfx1250) — matrix operand order repaired and checked by reading, not compiled, not run on hardware
//
#include <hip/hip_runtime.h>

// ---------------- problem constants (from reference) ----------------
constexpr int VOC = 32000;   // vocab
constexpr int H   = 1024;    // hidden
constexpr int T   = 32;      // time steps
constexpr int NB  = 64;      // batch
constexpr int SOS = 1;
constexpr int EOS = 2;
constexpr int H4  = 4 * H;

// ---------------- types ----------------
typedef __bf16 bf16;
typedef __attribute__((ext_vector_type(16))) __bf16 v16bf;
typedef __attribute__((ext_vector_type(8)))  __bf16 bf16x8;
typedef __attribute__((ext_vector_type(8)))  float  v8f;

union FragBF { v16bf v; bf16x8 h[2]; };

#define DEV __device__ __forceinline__

DEV float sigmoidf_(float x) { return 1.0f / (1.0f + __expf(-x)); }

// WMMA bf16 16x16x32, fp32 accumulate
DEV v8f wmma_bf16(v16bf a, v16bf b, v8f c) {
  return __builtin_amdgcn_wmma_f32_16x16x32_bf16(
      /*neg_a=*/false, a, /*neg_b=*/false, b,
      /*c_mod=*/(short)0, c, /*reuse_a=*/false, /*reuse_b=*/false);
}

// A fragment: 16x32 bf16 tile of row-major X[*, ldk], rows mtile*16..+15, cols kbase..kbase+31.
// ISA layout: lanes 0-15 hold M=lane, K = {0..7, 16..23}; lanes 16-31 hold K = {8..15, 24..31}.
DEV v16bf load_A(const bf16* __restrict__ X, int mtile, int kbase, int ldk, int lane) {
  int half = lane >> 4, mrow = lane & 15;
  const bf16* row = X + (size_t)(mtile * 16 + mrow) * ldk;
  FragBF f;
  f.h[0] = *(const bf16x8*)(row + kbase + 8 * half);
  f.h[1] = *(const bf16x8*)(row + kbase + 16 + 8 * half);
  return f.v;
}

// B fragment: 32x16 bf16 tile of W^T where W is row-major [N, ldk].
// B[k][n] = W[nbase + n][kbase + k]. Lane = column n (mod 16); lanes 0-15 hold K = {0..7, 16..23}; lanes 16-31 hold K = {8..15, 24..31}.
DEV v16bf load_B(const bf16* __restrict__ W, int nbase, int kbase, int ldk, int lane) {
  int khalf = lane >> 4, ncol = lane & 15;
  const bf16* row = W + (size_t)(nbase + ncol) * ldk;
  FragBF f;
  f.h[0] = *(const bf16x8*)(row + kbase + 8 * khalf);
  f.h[1] = *(const bf16x8*)(row + kbase + 16 + 8 * khalf);
  return f.v;
}

// ---------------- kernels ----------------

__global__ __launch_bounds__(256)
void k_convert(const float* __restrict__ src, bf16* __restrict__ dst, int n) {
  int stride = gridDim.x * blockDim.x;
  for (int i = blockIdx.x * blockDim.x + threadIdx.x; i < n; i += stride)
    dst[i] = (bf16)src[i];
}

__global__ __launch_bounds__(256)
void k_init(const float* __restrict__ E, const float* __restrict__ ehid,
            const float* __restrict__ ecell,
            bf16* __restrict__ xbf, bf16* __restrict__ hbf, float* __restrict__ c) {
  int i = blockIdx.x * blockDim.x + threadIdx.x;
  if (i >= NB * H) return;
  int j = i & (H - 1);
  xbf[i] = (bf16)E[(size_t)SOS * H + j];   // x0 = E[SOS] broadcast over batch
  hbf[i] = (bf16)ehid[i];
  c[i]   = ecell[i];
}

// Software-pipelined X[mtile-rows] @ W^T into 4 gate accumulators, 2x-unrolled
// ping-pong buffers (no rotation copies): stage loads write directly into the
// buffer set that the *next* WMMA group consumes, so the scheduler issues
// partial s_wait_loadcnt and keeps loads in flight under the matrix pipe.
DEV void gemm_gates_pipe(const bf16* __restrict__ X, int mtile,
                         const bf16* __restrict__ W, int nbase,
                         v8f acc[4], int lane) {
  v16bf a0 = load_A(X, mtile, 0, H, lane);
  v16bf b0[4];
#pragma unroll
  for (int g = 0; g < 4; ++g) b0[g] = load_B(W, g * H + nbase, 0, H, lane);

  for (int kb = 0; kb < H - 64; kb += 64) {
    v16bf a1 = load_A(X, mtile, kb + 32, H, lane);
    v16bf b1[4];
#pragma unroll
    for (int g = 0; g < 4; ++g) b1[g] = load_B(W, g * H + nbase, kb + 32, H, lane);
#pragma unroll
    for (int g = 0; g < 4; ++g) acc[g] = wmma_bf16(a0, b0[g], acc[g]);

    a0 = load_A(X, mtile, kb + 64, H, lane);
#pragma unroll
    for (int g = 0; g < 4; ++g) b0[g] = load_B(W, g * H + nbase, kb + 64, H, lane);
#pragma unroll
    for (int g = 0; g < 4; ++g) acc[g] = wmma_bf16(a1, b1[g], acc[g]);
  }
  // last two k-steps: H-64 is in (a0,b0); load H-32 into (a1,b1)
  {
    v16bf a1 = load_A(X, mtile, H - 32, H, lane);
    v16bf b1[4];
#pragma unroll
    for (int g = 0; g < 4; ++g) b1[g] = load_B(W, g * H + nbase, H - 32, H, lane);
#pragma unroll
    for (int g = 0; g < 4; ++g) acc[g] = wmma_bf16(a0, b0[g], acc[g]);
#pragma unroll
    for (int g = 0; g < 4; ++g) acc[g] = wmma_bf16(a1, b1[g], acc[g]);
  }
}

// gates = x@Wih^T + h@Whh^T + b_ih + b_hh; fused LSTM cell -> c (fp32), h_out (bf16)
// 64 waves: wave = mtile*64 + ntile; each wave: one 16x16 hidden tile, 4 gate accumulators.
__global__ __launch_bounds__(256)
void k_gates(const bf16* __restrict__ xbf, const bf16* __restrict__ hin,
             const bf16* __restrict__ Wih, const bf16* __restrict__ Whh,
             const float* __restrict__ bih, const float* __restrict__ bhh,
             float* __restrict__ c, bf16* __restrict__ hout) {
  int lane  = threadIdx.x & 31;
  int wave  = (blockIdx.x * blockDim.x + threadIdx.x) >> 5; // 0..63
  int ntile = wave & 63;   // H/16
  int mtile = wave >> 6;   // NB/16
  int nbase = ntile * 16;

  v8f acc[4] = {};
  gemm_gates_pipe(xbf, mtile, Wih, nbase, acc, lane);
  gemm_gates_pipe(hin, mtile, Whh, nbase, acc, lane);

  int half = lane >> 4, nlo = lane & 15;
  int col  = nbase + nlo;
  float bi  = bih[0 * H + col] + bhh[0 * H + col];
  float bf_ = bih[1 * H + col] + bhh[1 * H + col];
  float bg  = bih[2 * H + col] + bhh[2 * H + col];
  float bo  = bih[3 * H + col] + bhh[3 * H + col];
#pragma unroll
  for (int r = 0; r < 8; ++r) {
    int m = mtile * 16 + r + 8 * half;          // D-matrix row mapping
    size_t idx = (size_t)m * H + col;
    float iv = acc[0][r] + bi;
    float fv = acc[1][r] + bf_;
    float gv = acc[2][r] + bg;
    float ov = acc[3][r] + bo;
    float cn = sigmoidf_(fv) * c[idx] + sigmoidf_(iv) * tanhf(gv);
    float hn = sigmoidf_(ov) * tanhf(cn);
    c[idx]    = cn;
    hout[idx] = (bf16)hn;
  }
}

// logits[t] = h@Wout^T + b_out ; 2000 waves, wave = ntile.
// B fragment shared across 4 M-tiles (W_out read exactly once per step);
// 2x-unrolled ping-pong pipeline: next stage's 5 loads overlap current 4 WMMAs.
__global__ __launch_bounds__(256)
void k_logits(const bf16* __restrict__ hbf, const bf16* __restrict__ Wout,
              const float* __restrict__ bout, float* __restrict__ out, int t) {
  int lane  = threadIdx.x & 31;
  int ntile = (blockIdx.x * blockDim.x + threadIdx.x) >> 5; // 0..1999
  int nbase = ntile * 16;

  v8f acc[4] = {};
  v16bf b0 = load_B(Wout, nbase, 0, H, lane);
  v16bf a0[4];
#pragma unroll
  for (int mt = 0; mt < 4; ++mt) a0[mt] = load_A(hbf, mt, 0, H, lane);

  for (int kb = 0; kb < H - 64; kb += 64) {
    v16bf b1 = load_B(Wout, nbase, kb + 32, H, lane);
    v16bf a1[4];
#pragma unroll
    for (int mt = 0; mt < 4; ++mt) a1[mt] = load_A(hbf, mt, kb + 32, H, lane);
#pragma unroll
    for (int mt = 0; mt < 4; ++mt) acc[mt] = wmma_bf16(a0[mt], b0, acc[mt]);

    b0 = load_B(Wout, nbase, kb + 64, H, lane);
#pragma unroll
    for (int mt = 0; mt < 4; ++mt) a0[mt] = load_A(hbf, mt, kb + 64, H, lane);
#pragma unroll
    for (int mt = 0; mt < 4; ++mt) acc[mt] = wmma_bf16(a1[mt], b1, acc[mt]);
  }
  {
    v16bf b1 = load_B(Wout, nbase, H - 32, H, lane);
    v16bf a1[4];
#pragma unroll
    for (int mt = 0; mt < 4; ++mt) a1[mt] = load_A(hbf, mt, H - 32, H, lane);
#pragma unroll
    for (int mt = 0; mt < 4; ++mt) acc[mt] = wmma_bf16(a0[mt], b0, acc[mt]);
#pragma unroll
    for (int mt = 0; mt < 4; ++mt) acc[mt] = wmma_bf16(a1[mt], b1, acc[mt]);
  }

  int half = lane >> 4, nlo = lane & 15;
  int n = nbase + nlo;
  float bias = bout[n];
  size_t base = (size_t)t * NB * VOC;
#pragma unroll
  for (int mt = 0; mt < 4; ++mt)
#pragma unroll
    for (int r = 0; r < 8; ++r) {
      int m = mt * 16 + r + 8 * half;
      out[base + (size_t)m * VOC + n] = acc[mt][r] + bias;
    }
}

// One workgroup per batch row: online softmax (max,sum) + first-index argmax over V,
// write mask = softmax[EOS], gather x_next = bf16(E[argmax]).
__global__ __launch_bounds__(256)
void k_reduce(float* __restrict__ out, const float* __restrict__ E,
              bf16* __restrict__ xbf, int t) {
  __shared__ float smax[256], ssum[256], svmax[256];
  __shared__ int   sidx[256];
  int row = blockIdx.x, tid = threadIdx.x;
  const float* lrow = out + (size_t)t * NB * VOC + (size_t)row * VOC;

  float m = -INFINITY, s = 0.0f, bm = -INFINITY;
  int bi = 0;
  for (int colv = tid; colv < VOC; colv += 256) {
    float v = lrow[colv];
    if (v > m) { s = s * __expf(m - v) + 1.0f; m = v; }
    else       { s += __expf(v - m); }
    if (v > bm) { bm = v; bi = colv; }
  }
  smax[tid] = m; ssum[tid] = s; svmax[tid] = bm; sidx[tid] = bi;
  __syncthreads();
  for (int off = 128; off > 0; off >>= 1) {
    if (tid < off) {
      float m1 = smax[tid], m2 = smax[tid + off];
      float M  = fmaxf(m1, m2);
      ssum[tid] = ssum[tid] * __expf(m1 - M) + ssum[tid + off] * __expf(m2 - M);
      smax[tid] = M;
      float v2 = svmax[tid + off]; int i2 = sidx[tid + off];
      if (v2 > svmax[tid] || (v2 == svmax[tid] && i2 < sidx[tid])) {
        svmax[tid] = v2; sidx[tid] = i2;
      }
    }
    __syncthreads();
  }
  if (tid == 0) {
    // masks live after outputs: offset T*NB*VOC + t*NB + row
    out[(size_t)T * NB * VOC + (size_t)t * NB + row] =
        __expf(lrow[EOS] - smax[0]) / ssum[0];
  }
  __syncthreads();
  int arg = sidx[0];
  const float* erow = E + (size_t)arg * H;
  bf16* xr = xbf + (size_t)row * H;
  for (int j = tid; j < H; j += 256) xr[j] = (bf16)erow[j];
}

// ---------------- workspace layout (bytes, 256-aligned) ----------------
constexpr size_t OFF_WIH  = 0;
constexpr size_t OFF_WHH  = OFF_WIH  + (size_t)H4 * H * 2;    //  8,388,608
constexpr size_t OFF_WOUT = OFF_WHH  + (size_t)H4 * H * 2;    // 16,777,216
constexpr size_t OFF_X    = OFF_WOUT + (size_t)VOC * H * 2;   // 82,313,216
constexpr size_t OFF_HA   = OFF_X    + (size_t)NB * H * 2;
constexpr size_t OFF_HB   = OFF_HA   + (size_t)NB * H * 2;
constexpr size_t OFF_C    = OFF_HB   + (size_t)NB * H * 2;
constexpr size_t WS_NEED  = OFF_C    + (size_t)NB * H * 4;    // ~83 MB

extern "C" void kernel_launch(void* const* d_in, const int* in_sizes, int n_in,
                              void* d_out, int out_size, void* d_ws, size_t ws_size,
                              hipStream_t stream) {
  (void)in_sizes; (void)n_in; (void)out_size;
  const float* E     = (const float*)d_in[0];
  const float* Wih   = (const float*)d_in[1];
  const float* Whh   = (const float*)d_in[2];
  const float* bih   = (const float*)d_in[3];
  const float* bhh   = (const float*)d_in[4];
  const float* Wout  = (const float*)d_in[5];
  const float* bout  = (const float*)d_in[6];
  const float* ehid  = (const float*)d_in[7];
  const float* ecell = (const float*)d_in[8];
  float* out = (float*)d_out;

  if (ws_size < WS_NEED) return;  // workspace too small -> cannot run this plan

  char* ws = (char*)d_ws;
  bf16*  wIhB  = (bf16*)(ws + OFF_WIH);
  bf16*  wHhB  = (bf16*)(ws + OFF_WHH);
  bf16*  wOutB = (bf16*)(ws + OFF_WOUT);
  bf16*  xbf   = (bf16*)(ws + OFF_X);
  bf16*  hA    = (bf16*)(ws + OFF_HA);
  bf16*  hB    = (bf16*)(ws + OFF_HB);
  float* c     = (float*)(ws + OFF_C);

  // one-time (per call) weight down-conversion to bf16 in workspace
  k_convert<<<1024, 256, 0, stream>>>(Wih,  wIhB,  H4 * H);
  k_convert<<<1024, 256, 0, stream>>>(Whh,  wHhB,  H4 * H);
  k_convert<<<4096, 256, 0, stream>>>(Wout, wOutB, VOC * H);
  k_init<<<(NB * H) / 256, 256, 0, stream>>>(E, ehid, ecell, xbf, hA, c);

  bf16* hin = hA;
  bf16* hout = hB;
  for (int t = 0; t < T; ++t) {
    k_gates<<<8,   256, 0, stream>>>(xbf, hin, wIhB, wHhB, bih, bhh, c, hout);
    k_logits<<<250, 256, 0, stream>>>(hout, wOutB, bout, out, t);
    k_reduce<<<NB, 256, 0, stream>>>(out, E, xbf, t);
    bf16* tmp = hin; hin = hout; hout = tmp;  // h double-buffer swap
  }
}
